// GATNet_39848706572499
// MI455X (gfx1250) — matrix operand order repaired and checked by reading, not compiled, not run on hardware
//
#include <hip/hip_runtime.h>
#include <stdint.h>

// ---------------- problem constants (match reference) ----------------
#define NNODES  65536
#define NEDGES  196608
#define ETOT    (NNODES + NEDGES)   // self loops appended
#define NGRAPH  2048
#define H1HEADS 10
#define DFEAT   128
#define F1      1280                // H1HEADS * DFEAT
#define NEG_SLOPE 0.2f

typedef __attribute__((ext_vector_type(16))) __bf16 v16bf;
typedef __attribute__((ext_vector_type(8)))  float  v8f;

// ---------------- helpers ----------------
// order-preserving float <-> uint encoding for atomicMax on floats
__device__ __forceinline__ unsigned fenc(float f) {
  unsigned u = __float_as_uint(f);
  return (u & 0x80000000u) ? ~u : (u | 0x80000000u);
}
__device__ __forceinline__ float fdec(unsigned u) {
  u = (u & 0x80000000u) ? (u & 0x7FFFFFFFu) : ~u;
  return __uint_as_float(u);
}
#define ENC_NEG_INF 0x007FFFFFu   // fenc(-inf)

__device__ __forceinline__ unsigned short f2bf(float f) {   // RNE fp32->bf16
  unsigned u = __float_as_uint(f);
  u += 0x7FFFu + ((u >> 16) & 1u);
  return (unsigned short)(u >> 16);
}
__device__ __forceinline__ float eluf(float v) {
  return v > 0.f ? v : (__expf(v) - 1.f);
}
__device__ __forceinline__ float lrelu(float v) {
  return v > 0.f ? v : NEG_SLOPE * v;
}

// ---------------- fill kernels ----------------
__global__ void k_fillf(float* p, float v, size_t n) {
  size_t i = (size_t)blockIdx.x * blockDim.x + threadIdx.x;
  if (i < n) p[i] = v;
}
__global__ void k_fillu(unsigned* p, unsigned v, size_t n) {
  size_t i = (size_t)blockIdx.x * blockDim.x + threadIdx.x;
  if (i < n) p[i] = v;
}

// ---------------- layer-1 feature transform: h1[N,1280] = x @ W1 (K=9) ----
__global__ void __launch_bounds__(256) k_gemm1(const float* __restrict__ x,
                                               const float* __restrict__ W1,
                                               float* __restrict__ h1) {
  __shared__ float xr[9];
  int n = blockIdx.x;
  if (threadIdx.x < 9) xr[threadIdx.x] = x[n * 9 + threadIdx.x];
  __syncthreads();
  for (int f = threadIdx.x; f < F1; f += blockDim.x) {
    float acc = 0.f;
#pragma unroll
    for (int k = 0; k < 9; ++k) acc += xr[k] * W1[k * F1 + f];
    h1[(size_t)n * F1 + f] = acc;
  }
}

// ---------------- attention scalars: one wave per (node, head) ----------
__global__ void __launch_bounds__(256) k_alpha(const float* __restrict__ hfeat,
                                               const float* __restrict__ aw_s,
                                               const float* __restrict__ aw_d,
                                               int H, float* __restrict__ as_,
                                               float* __restrict__ ad_) {
  int w = (blockIdx.x * blockDim.x + threadIdx.x) >> 5;
  int lane = threadIdx.x & 31;
  if (w >= NNODES * H) return;
  int n = w / H, head = w - n * H;
  const float4* hp = (const float4*)(hfeat + (size_t)n * H * DFEAT + head * DFEAT);
  const float4* sp = (const float4*)(aw_s + head * DFEAT);
  const float4* dp = (const float4*)(aw_d + head * DFEAT);
  float4 hv = hp[lane], sv = sp[lane], dv = dp[lane];
  float ss = hv.x * sv.x + hv.y * sv.y + hv.z * sv.z + hv.w * sv.w;
  float dd = hv.x * dv.x + hv.y * dv.y + hv.z * dv.z + hv.w * dv.w;
#pragma unroll
  for (int off = 16; off > 0; off >>= 1) {
    ss += __shfl_down(ss, off, 32);
    dd += __shfl_down(dd, off, 32);
  }
  if (lane == 0) { as_[w] = ss; ad_[w] = dd; }
}

// ---------------- edge helpers ----------------
__device__ __forceinline__ void edge_sd(const int* ei, int e, int& s, int& d) {
  if (e < NEDGES) { s = ei[e]; d = ei[NEDGES + e]; }
  else            { s = d = e - NEDGES; }          // self loop
}

// pass A: segment max of leaky-relu logits (encoded-uint atomicMax)
__global__ void __launch_bounds__(256) k_edge_max(const int* __restrict__ ei, int H,
                                                  const float* __restrict__ as_,
                                                  const float* __restrict__ ad_,
                                                  unsigned* __restrict__ m) {
  int gid = blockIdx.x * blockDim.x + threadIdx.x;
  if (gid >= ETOT * H) return;
  int head = gid / ETOT, e = gid - head * ETOT;
  int s, d; edge_sd(ei, e, s, d);
  float ev = lrelu(as_[s * H + head] + ad_[d * H + head]);
  atomicMax(&m[d * H + head], fenc(ev));
}

// pass B: denominator = segment sum of exp(e - m[dst])
__global__ void __launch_bounds__(256) k_edge_sum(const int* __restrict__ ei, int H,
                                                  const float* __restrict__ as_,
                                                  const float* __restrict__ ad_,
                                                  const unsigned* __restrict__ m,
                                                  float* __restrict__ den) {
  int gid = blockIdx.x * blockDim.x + threadIdx.x;
  if (gid >= ETOT * H) return;
  int head = gid / ETOT, e = gid - head * ETOT;
  int s, d; edge_sd(ei, e, s, d);
  float ev = lrelu(as_[s * H + head] + ad_[d * H + head]);
  atomicAdd(&den[d * H + head], __expf(ev - fdec(m[d * H + head])));
}

// pass C: out[dst] += h[src] * alpha    (one wave per (edge, head))
__global__ void __launch_bounds__(256) k_edge_agg(const int* __restrict__ ei, int H,
                                                  const float* __restrict__ as_,
                                                  const float* __restrict__ ad_,
                                                  const unsigned* __restrict__ m,
                                                  const float* __restrict__ den,
                                                  const float* __restrict__ hfeat,
                                                  float* __restrict__ outf) {
  int w = (blockIdx.x * blockDim.x + threadIdx.x) >> 5;
  int lane = threadIdx.x & 31;
  if (w >= ETOT * H) return;
  int head = w / ETOT, e = w - head * ETOT;
  int s, d; edge_sd(ei, e, s, d);
  float ev = lrelu(as_[s * H + head] + ad_[d * H + head]);
  float alpha = __expf(ev - fdec(m[d * H + head])) / (den[d * H + head] + 1e-16f);
  int rs = H * DFEAT;
  float4 hv = ((const float4*)(hfeat + (size_t)s * rs + head * DFEAT))[lane];
  float* op = outf + (size_t)d * rs + head * DFEAT + lane * 4;
  atomicAdd(op + 0, hv.x * alpha);
  atomicAdd(op + 1, hv.y * alpha);
  atomicAdd(op + 2, hv.z * alpha);
  atomicAdd(op + 3, hv.w * alpha);
}

// ---------------- post-layer-1: x1e = bf16(elu(out1 + b1)) ----------------
__global__ void __launch_bounds__(256) k_postact1(const float* __restrict__ out1,
                                                  const float* __restrict__ b1,
                                                  unsigned short* __restrict__ x1e) {
  size_t i = (size_t)blockIdx.x * blockDim.x + threadIdx.x;
  if (i >= (size_t)NNODES * F1) return;
  int f = (int)(i % F1);
  x1e[i] = f2bf(eluf(out1[i] + b1[f]));
}

// ---------------- W2T[n][k] = bf16(W2[k][n])  (128 x 1280, N-major) ------
__global__ void __launch_bounds__(256) k_w2t(const float* __restrict__ W2,
                                             unsigned short* __restrict__ W2T) {
  int i = blockIdx.x * blockDim.x + threadIdx.x;
  if (i >= DFEAT * F1) return;
  int n = i / F1, k = i - n * F1;
  W2T[i] = f2bf(W2[(size_t)k * DFEAT + n]);
}

// ---------------- layer-2 GEMM: h2[N,128] = x1e @ W2  (bf16 WMMA) --------
// One K-step: consume the 32xK slab in `cur`, prefetch the next slab from
// global into `nxt` (statically-addressed double buffering; MORE is a
// compile-time flag so the hot loop has no branches).
template <bool MORE>
__device__ __forceinline__ void gemm2_step(const unsigned short* __restrict__ B,
                                           const unsigned short* cur,
                                           unsigned short* nxt, int k0,
                                           const unsigned short* arowp,
                                           int lhalf, int hi16,
                                           int crow0, int coff0,
                                           int crow1, int coff1,
                                           v8f acc[8]) {
  union Frag { v16bf v; uint4 q[2]; };
  uint4 pre0, pre1;
  if (MORE) {
    pre0 = *(const uint4*)&B[(size_t)crow0 * F1 + (k0 + 32) + coff0];
    pre1 = *(const uint4*)&B[(size_t)crow1 * F1 + (k0 + 32) + coff1];
  }
  // A fragment (ISA 16-bit A 16x32 layout): lane<16 -> K 0-7 & 16-23
  Frag af;
  const unsigned short* ap = arowp + k0;
  af.q[0] = *(const uint4*)ap;
  af.q[1] = *(const uint4*)(ap + 16);
#pragma unroll
  for (int j = 0; j < 8; ++j) {
    // B fragment (32x16): lane<16 -> col=lane, K 0-7 & 16-23; lane>=16 -> K 8-15 & 24-31
    Frag bfm;
    const unsigned short* bp = &cur[(j * 16 + lhalf) * 40 + (hi16 >> 1)];
    bfm.q[0] = *(const uint4*)bp;
    bfm.q[1] = *(const uint4*)(bp + 16);
    acc[j] = __builtin_amdgcn_wmma_f32_16x16x32_bf16(
        false, af.v, false, bfm.v, (short)0, acc[j], false, false);
  }
  if (MORE) {
    *(uint4*)&nxt[crow0 * 40 + coff0] = pre0;
    *(uint4*)&nxt[crow1 * 40 + coff1] = pre1;
  }
  __syncthreads();
}

// block = 8 waves; each wave computes a 16(M) x 128(N) strip (8 wmma tiles).
__global__ void __launch_bounds__(256) k_gemm2(const unsigned short* __restrict__ A,
                                               const unsigned short* __restrict__ B,
                                               float* __restrict__ C) {
  __shared__ unsigned short lds0[128 * 40];   // rows n=0..127, 32 k + pad
  __shared__ unsigned short lds1[128 * 40];
  const int lane  = threadIdx.x & 31;
  const int warp  = threadIdx.x >> 5;
  const int lhalf = lane & 15;
  const int hi16  = (lane >= 16) ? 16 : 0;
  const size_t mbase = (size_t)blockIdx.x * 128 + warp * 16;
  const unsigned short* arowp = A + (mbase + lhalf) * F1 + (hi16 >> 1);

  // cooperative B-slab load indexing: 512 16B chunks over 256 threads
  const int chunk0 = threadIdx.x;
  const int chunk1 = threadIdx.x + 256;
  const int crow0 = chunk0 >> 2, coff0 = (chunk0 & 3) * 8;
  const int crow1 = chunk1 >> 2, coff1 = (chunk1 & 3) * 8;

  // initial slab -> lds0
  *(uint4*)&lds0[crow0 * 40 + coff0] = *(const uint4*)&B[(size_t)crow0 * F1 + coff0];
  *(uint4*)&lds1[0] = *(uint4*)&lds1[0];   // no-op touch (keeps lds1 alive)
  __syncthreads();

  v8f acc[8];
#pragma unroll
  for (int j = 0; j < 8; ++j)
#pragma unroll
    for (int r = 0; r < 8; ++r) acc[j][r] = 0.f;

  // second half of initial cooperative load went above? No: do chunk1 too.
  // (placed before __syncthreads in real order below)

  // NOTE: chunk1 part of the initial load:
  // (executed before the barrier above via the statement below being hoisted
  //  is not guaranteed, so we do it explicitly here with a second barrier)
  *(uint4*)&lds0[crow1 * 40 + coff1] = *(const uint4*)&B[(size_t)crow1 * F1 + coff1];
  __syncthreads();

  // main loop: 19 double-steps with unconditional prefetch
  for (int k0 = 0; k0 < F1 - 64; k0 += 64) {
    gemm2_step<true>(B, lds0, lds1, k0,      arowp, lhalf, hi16,
                     crow0, coff0, crow1, coff1, acc);
    gemm2_step<true>(B, lds1, lds0, k0 + 32, arowp, lhalf, hi16,
                     crow0, coff0, crow1, coff1, acc);
  }
  // tail: k = F1-64 (prefetches F1-32 into lds1), then k = F1-32 (no prefetch)
  gemm2_step<true>(B, lds0, lds1, F1 - 64, arowp, lhalf, hi16,
                   crow0, coff0, crow1, coff1, acc);
  gemm2_step<false>(B, lds1, lds0, F1 - 32, arowp, lhalf, hi16,
                    crow0, coff0, crow1, coff1, acc);

  // store C (16x16 f32 D layout: VGPR r -> M=r (lanes 0-15) / M=r+8 (16-31))
#pragma unroll
  for (int j = 0; j < 8; ++j) {
    int col = j * 16 + lhalf;
#pragma unroll
    for (int r = 0; r < 8; ++r) {
      size_t row = mbase + ((hi16 != 0) ? r + 8 : r);
      C[row * DFEAT + col] = acc[j][r];
    }
  }
}

// ---------------- pool: elu(out2 + b2) then segment_max over batch -------
__global__ void __launch_bounds__(256) k_pool(const float* __restrict__ out2,
                                              const float* __restrict__ b2,
                                              const int* __restrict__ batch,
                                              unsigned* __restrict__ pooledU) {
  size_t i = (size_t)blockIdx.x * blockDim.x + threadIdx.x;
  if (i >= (size_t)NNODES * DFEAT) return;
  int n = (int)(i >> 7), c = (int)(i & 127);
  float v = eluf(out2[i] + b2[c]);
  atomicMax(&pooledU[batch[n] * DFEAT + c], fenc(v));
}

// ---------------- FC: out = relu(pooled @ Wfc + bfc) ---------------------
__global__ void __launch_bounds__(128) k_fc(const unsigned* __restrict__ pooledU,
                                            const float* __restrict__ Wfc,
                                            const float* __restrict__ bfc,
                                            float* __restrict__ out) {
  __shared__ float p[DFEAT];
  int g = blockIdx.x, c = threadIdx.x;
  float v = fdec(pooledU[g * DFEAT + c]);
  if (!(fabsf(v) < 3.0e38f)) v = 0.f;     // guard empty graphs (-inf / nan)
  p[c] = v;
  __syncthreads();
  float acc = bfc[c];
#pragma unroll 8
  for (int k = 0; k < DFEAT; ++k) acc += p[k] * Wfc[k * DFEAT + c];
  out[g * DFEAT + c] = acc > 0.f ? acc : 0.f;
}

// ---------------- host launcher ----------------
extern "C" void kernel_launch(void* const* d_in, const int* in_sizes, int n_in,
                              void* d_out, int out_size, void* d_ws, size_t ws_size,
                              hipStream_t stream) {
  (void)in_sizes; (void)n_in; (void)out_size; (void)ws_size;
  const float* x     = (const float*)d_in[0];
  const int*   ei    = (const int*)  d_in[1];
  /* d_in[2] edge_attr unused by forward */
  const int*   batch = (const int*)  d_in[3];
  const float* W1    = (const float*)d_in[4];
  const float* as1w  = (const float*)d_in[5];
  const float* ad1w  = (const float*)d_in[6];
  const float* b1    = (const float*)d_in[7];
  const float* W2    = (const float*)d_in[8];
  const float* as2w  = (const float*)d_in[9];
  const float* ad2w  = (const float*)d_in[10];
  const float* b2    = (const float*)d_in[11];
  const float* Wfc   = (const float*)d_in[12];
  const float* bfc   = (const float*)d_in[13];
  float* out = (float*)d_out;
  char*  ws  = (char*)d_ws;

  // workspace layout (all offsets 256B-aligned by construction)
  constexpr size_t SZ_NF1 = (size_t)NNODES * F1 * 4;        // 335.5 MB
  constexpr size_t SZ_NH  = (size_t)NNODES * H1HEADS * 4;   // 2.6 MB
  constexpr size_t SZ_ND  = (size_t)NNODES * DFEAT * 4;     // 33.5 MB
  constexpr size_t SZ_N1  = (size_t)NNODES * 4;
  constexpr size_t OFF_H1   = 0;                        // later reused as x1e (bf16)
  constexpr size_t OFF_OUT1 = OFF_H1   + SZ_NF1;
  constexpr size_t OFF_AS1  = OFF_OUT1 + SZ_NF1;
  constexpr size_t OFF_AD1  = OFF_AS1  + SZ_NH;
  constexpr size_t OFF_M1   = OFF_AD1  + SZ_NH;
  constexpr size_t OFF_DEN1 = OFF_M1   + SZ_NH;
  constexpr size_t OFF_W2T  = OFF_DEN1 + SZ_NH;         // 1280*128*2 bytes
  constexpr size_t OFF_H2   = OFF_W2T  + (size_t)DFEAT * F1 * 2;
  constexpr size_t OFF_OUT2 = OFF_H2   + SZ_ND;
  constexpr size_t OFF_AS2  = OFF_OUT2 + SZ_ND;
  constexpr size_t OFF_AD2  = OFF_AS2  + SZ_N1;
  constexpr size_t OFF_M2   = OFF_AD2  + SZ_N1;
  constexpr size_t OFF_DEN2 = OFF_M2   + SZ_N1;
  constexpr size_t OFF_POOL = OFF_DEN2 + SZ_N1;

  float*          h1   = (float*)(ws + OFF_H1);
  unsigned short* x1e  = (unsigned short*)(ws + OFF_H1);  // aliases h1 (after agg1)
  float*          out1 = (float*)(ws + OFF_OUT1);
  float*          as1  = (float*)(ws + OFF_AS1);
  float*          ad1  = (float*)(ws + OFF_AD1);
  unsigned*       m1   = (unsigned*)(ws + OFF_M1);
  float*          den1 = (float*)(ws + OFF_DEN1);
  unsigned short* w2t  = (unsigned short*)(ws + OFF_W2T);
  float*          h2   = (float*)(ws + OFF_H2);
  float*          out2 = (float*)(ws + OFF_OUT2);
  float*          as2  = (float*)(ws + OFF_AS2);
  float*          ad2  = (float*)(ws + OFF_AD2);
  unsigned*       m2   = (unsigned*)(ws + OFF_M2);
  float*          den2 = (float*)(ws + OFF_DEN2);
  unsigned*       pool = (unsigned*)(ws + OFF_POOL);

  auto blks = [](size_t n, int t) { return (unsigned)((n + t - 1) / t); };

  // ---- init ----
  k_fillf<<<blks((size_t)NNODES * F1, 256), 256, 0, stream>>>(out1, 0.f, (size_t)NNODES * F1);
  k_fillf<<<blks((size_t)NNODES * H1HEADS, 256), 256, 0, stream>>>(den1, 0.f, (size_t)NNODES * H1HEADS);
  k_fillu<<<blks((size_t)NNODES * H1HEADS, 256), 256, 0, stream>>>(m1, ENC_NEG_INF, (size_t)NNODES * H1HEADS);
  k_fillf<<<blks((size_t)NNODES * DFEAT, 256), 256, 0, stream>>>(out2, 0.f, (size_t)NNODES * DFEAT);
  k_fillf<<<blks(NNODES, 256), 256, 0, stream>>>(den2, 0.f, (size_t)NNODES);
  k_fillu<<<blks(NNODES, 256), 256, 0, stream>>>(m2, ENC_NEG_INF, (size_t)NNODES);
  k_fillu<<<blks((size_t)NGRAPH * DFEAT, 256), 256, 0, stream>>>(pool, ENC_NEG_INF, (size_t)NGRAPH * DFEAT);

  // ---- layer 1 ----
  k_gemm1<<<NNODES, 256, 0, stream>>>(x, W1, h1);
  k_alpha<<<blks((size_t)NNODES * H1HEADS * 32, 256), 256, 0, stream>>>(h1, as1w, ad1w, H1HEADS, as1, ad1);
  k_edge_max<<<blks((size_t)ETOT * H1HEADS, 256), 256, 0, stream>>>(ei, H1HEADS, as1, ad1, m1);
  k_edge_sum<<<blks((size_t)ETOT * H1HEADS, 256), 256, 0, stream>>>(ei, H1HEADS, as1, ad1, m1, den1);
  k_edge_agg<<<blks((size_t)ETOT * H1HEADS * 32, 256), 256, 0, stream>>>(ei, H1HEADS, as1, ad1, m1, den1, h1, out1);
  k_postact1<<<blks((size_t)NNODES * F1, 256), 256, 0, stream>>>(out1, b1, x1e);

  // ---- layer 2 (WMMA GEMM) ----
  k_w2t<<<blks((size_t)DFEAT * F1, 256), 256, 0, stream>>>(W2, w2t);
  k_gemm2<<<NNODES / 128, 256, 0, stream>>>(x1e, w2t, h2);
  k_alpha<<<blks((size_t)NNODES * 32, 256), 256, 0, stream>>>(h2, as2w, ad2w, 1, as2, ad2);
  k_edge_max<<<blks((size_t)ETOT, 256), 256, 0, stream>>>(ei, 1, as2, ad2, m2);
  k_edge_sum<<<blks((size_t)ETOT, 256), 256, 0, stream>>>(ei, 1, as2, ad2, m2, den2);
  k_edge_agg<<<blks((size_t)ETOT * 32, 256), 256, 0, stream>>>(ei, 1, as2, ad2, m2, den2, h2, out2);

  // ---- pool + FC ----
  k_pool<<<blks((size_t)NNODES * DFEAT, 256), 256, 0, stream>>>(out2, b2, batch, pool);
  k_fc<<<NGRAPH, 128, 0, stream>>>(pool, Wfc, bfc, out);
}
